// SequenceEncoder_89730456748097
// MI455X (gfx1250) — matrix operand order repaired and checked by reading, not compiled, not run on hardware
//
#include <hip/hip_runtime.h>
#include <hip/hip_bf16.h>

// ---------------------------------------------------------------------------
// SequenceEncoder fused kernel for gfx1250 (MI455X, wave32, WMMA).
//
// N=256, L=4096, DIM_LINEAR=64, DIM_CONV=128, KERNEL=15 -> Lout=4082
//
//   prep_tables : W2[t*4+s][o] = sum_c conv_w[o,c,t]*emb[s,c]  (tokens in
//                 {0..3} collapse the conv to a 60x128 table);  repack W2,
//                 Wk, Wv into bf16 WMMA B-fragment lane order.
//   mean_q      : windowed token histogram -> read_mean -> q  (per sequence).
//   attn        : one block / sequence, 8 waves.  Per 32-position super-tile:
//                   y  = OneHot(tokens) @ W2    (4 WMMA / wave; one-hot from
//                                                two 32-bit lane bitmasks)
//                   k  = y @ Wk^T, v = y @ Wv^T (16 WMMA / wave)
//                   block-parallel scores + online-softmax accumulate.
//                 bk dropped (softmax-invariant), bv added once at the end.
// ---------------------------------------------------------------------------

#define SEQ_N      256
#define SEQ_L      4096
#define DL         64
#define DC         128
#define KW         15
#define LOUT       (SEQ_L - KW + 1)   // 4082
#define PTILE      32                 // positions per iteration
#define NITER      128                // 4096 / 32 (tail masked)

typedef __attribute__((ext_vector_type(16))) __bf16 v16bf;
typedef __attribute__((ext_vector_type(8)))  __bf16 v8bf;
typedef __attribute__((ext_vector_type(8)))  float  v8f;

// ws layout (bytes)
#define WS_W2    0          // float  [60][128]          30720 B
#define WS_Q     30720      // float  [256][128]        131072 B
#define WS_WKF   161792     // __bf16 [8][4][32][16]     32768 B
#define WS_WVF   194560     // __bf16 [8][4][32][16]     32768 B
#define WS_W2F   227328     // __bf16 [8][2][32][16]     16384 B

// ---------------------------------------------------------------------------
__global__ void prep_tables(const float* __restrict__ emb,
                            const float* __restrict__ conv_w,
                            const float* __restrict__ Wk,
                            const float* __restrict__ Wv,
                            float* __restrict__ W2,
                            __bf16* __restrict__ WkF,
                            __bf16* __restrict__ WvF,
                            __bf16* __restrict__ W2F) {
    int tid = threadIdx.x;
    // W2[(t*4+s)*128+o] = sum_c conv_w[o][c][t] * emb[s][c]
    for (int idx = tid; idx < 60 * DC; idx += blockDim.x) {
        int o = idx & 127;
        int ts = idx >> 7;          // t*4+s
        int t = ts >> 2, s = ts & 3;
        float acc = 0.f;
        for (int c = 0; c < DL; ++c)
            acc += conv_w[(o * DL + c) * KW + t] * emb[s * DL + c];
        W2[idx] = acc;
    }
    // B fragments for bf16 WMMA 16x16x32: B is 32x16 (KxN).
    // Lanes 0-15 hold K=kk*32+{0..7,16..23}, lanes 16-31 hold K=kk*32+{8..15,24..31},
    // column N = j*16 + (lane&15).  Projections use W^T: B[k][n] = W[n][k].
    for (int idx = tid; idx < 8 * 4 * 32; idx += blockDim.x) {
        int j = idx >> 7;               // output column group (wave id)
        int r = idx & 127;
        int kk = r >> 5, lane = r & 31;
        int col = j * 16 + (lane & 15);
        int kb = kk * 32 + ((lane >> 4) << 3);
        for (int e = 0; e < 16; ++e) {
            WkF[idx * 16 + e] = (__bf16)Wk[col * DC + kb + (e & 7) + ((e >> 3) << 4)];
            WvF[idx * 16 + e] = (__bf16)Wv[col * DC + kb + (e & 7) + ((e >> 3) << 4)];
        }
    }
    __syncthreads();                    // W2F reads W2 (single block)
    // W2 as B matrix (64x16 per column group, K = t*4+s padded 60->64)
    for (int idx = tid; idx < 8 * 2 * 32; idx += blockDim.x) {
        int j = idx >> 6;
        int r = idx & 63;
        int kk = r >> 5, lane = r & 31;
        int n = j * 16 + (lane & 15);
        int kb = kk * 32 + ((lane >> 4) << 3);
        for (int e = 0; e < 16; ++e) {
            int k = kb + (e & 7) + ((e >> 3) << 4);
            W2F[idx * 16 + e] = (k < 60) ? (__bf16)W2[k * DC + n] : (__bf16)0.f;
        }
    }
}

// ---------------------------------------------------------------------------
__global__ void mean_q(const int* __restrict__ tokens,
                       const float* __restrict__ conv_b,
                       const float* __restrict__ Wq,
                       const float* __restrict__ bq,
                       const float* __restrict__ W2,
                       float* __restrict__ qws) {
    __shared__ int   h[4];
    __shared__ int   cnt[KW][4];
    __shared__ float meanS[DC];
    int n = blockIdx.x, tid = threadIdx.x;
    const int* tk = tokens + n * SEQ_L;

    if (tid < 4) h[tid] = 0;
    __syncthreads();
    for (int l = tid; l < SEQ_L; l += blockDim.x)
        atomicAdd(&h[tk[l]], 1);
    __syncthreads();
    if (tid == 0) {
        // cnt[t][s] = #positions p in [t, t+LOUT) with token s
        int pre[4] = {0, 0, 0, 0}, post[4] = {0, 0, 0, 0};
        for (int p = LOUT; p < SEQ_L; ++p) post[tk[p]]++;
        for (int t = 0; t < KW; ++t) {
            for (int s = 0; s < 4; ++s) cnt[t][s] = h[s] - pre[s] - post[s];
            pre[tk[t]]++;
            if (t + LOUT < SEQ_L) post[tk[t + LOUT]]--;
        }
    }
    __syncthreads();
    // read_mean[o] = conv_b[o] + (1/Lout) * sum_{t,s} cnt[t][s]*W2[t][s][o]
    float acc = 0.f;
    for (int t = 0; t < KW; ++t)
        for (int s = 0; s < 4; ++s)
            acc += (float)cnt[t][s] * W2[(t * 4 + s) * DC + tid];
    meanS[tid] = conv_b[tid] + acc * (1.0f / (float)LOUT);
    __syncthreads();
    float qv = bq[tid];
    for (int d = 0; d < DC; ++d) qv += Wq[tid * DC + d] * meanS[d];
    qws[n * DC + tid] = qv;
}

// ---------------------------------------------------------------------------
__global__ void
__launch_bounds__(256)
attn(const int* __restrict__ tokens,
     const float* __restrict__ conv_b,
     const float* __restrict__ bv,
     const float* __restrict__ qws,
     const __bf16* __restrict__ WkF,
     const __bf16* __restrict__ WvF,
     const __bf16* __restrict__ W2F,
     float* __restrict__ out) {
    __shared__ float cb[DC], qS[DC], accS[DC];
    __shared__ float kT[PTILE * DC];                 // k tile, f32  (16 KB)
    __shared__ float vT[PTILE * DC];                 // v tile, f32  (16 KB)
    __shared__ __align__(16) __bf16 yA[PTILE * DC];  // y tile, bf16 ( 8 KB)
    __shared__ int   sTok[48];
    __shared__ float scP[PTILE * 16];                // score partials (2 KB)
    __shared__ float sc[PTILE], ev[PTILE];
    __shared__ float mZ[2];                          // running max, running Z
    __shared__ float scaleS, newmS;

    int n = blockIdx.x, tid = threadIdx.x;
    int lane = tid & 31, wid = tid >> 5;             // 8 waves per block

    if (tid < DC) {
        cb[tid]   = conv_b[tid];
        qS[tid]   = qws[n * DC + tid];
        accS[tid] = 0.f;
    }
    if (tid == 0) { mZ[0] = -1e30f; mZ[1] = 0.f; }

    // Per-wave B fragments, resident in VGPRs for the whole sequence.
    v16bf kB[4], vB[4], wB[2];
#pragma unroll
    for (int kk = 0; kk < 4; ++kk) {
        int fi = (wid * 4 + kk) * 32 + lane;
        kB[kk] = *((const v16bf*)(WkF + fi * 16));
        vB[kk] = *((const v16bf*)(WvF + fi * 16));
    }
#pragma unroll
    for (int kk = 0; kk < 2; ++kk) {
        int fi = (wid * 2 + kk) * 32 + lane;
        wB[kk] = *((const v16bf*)(W2F + fi * 16));
    }
    __syncthreads();

    const int* tk = tokens + n * SEQ_L;
    const float scl = 0.08838834764831845f;          // 1/sqrt(128)
    const __bf16 oneBF  = (__bf16)1.0f;
    const __bf16 zeroBF = (__bf16)0.0f;

    int m     = lane & 15;                           // A row within 16-group
    int rbase = (lane >> 4) * 8;                     // D row base
    int col   = wid * 16 + (lane & 15);              // D column (channel)
    int half  = (lane >> 4) * 8;                     // A K sub-block select

    float cbc = cb[col];                             // lane-constant conv bias
    float qc[8];                                     // q slice for score stage1
    {
        int c = tid & 15;
#pragma unroll
        for (int e = 0; e < 8; ++e) qc[e] = qS[c * 8 + e];
    }

    // one-hot A fragment from a 32-bit mask (K sub-chunk of 32)
    auto mkfrag = [&](unsigned msk) -> v16bf {
        v16bf a;
#pragma unroll
        for (int e = 0; e < 16; ++e) {
            int kidx = (e & 7) + ((e >> 3) * 16) + ((lane >> 4) * 8);
            a[e] = ((msk >> kidx) & 1u) ? oneBF : zeroBF;
        }
        return a;
    };

    for (int it = 0; it < NITER; ++it) {
        int l0 = it * PTILE;
        if (tid < 48) {
            int p = l0 + tid;
            sTok[tid] = tk[p < SEQ_L ? p : (SEQ_L - 1)];
            int pn = l0 + PTILE + tid;               // prefetch next window
            __builtin_prefetch(&tk[pn < SEQ_L ? pn : (SEQ_L - 1)], 0, 1);
        }
        __syncthreads();

        // ---- y super-tile via one-hot WMMA: y = OneHot(32x64) @ W2(64x128) -
        // lo mask: K=t*4+s for t=0..7; hi mask: t=8..14 (pads 60..63 zero)
#pragma unroll
        for (int hrow = 0; hrow < 2; ++hrow) {
            int mrow = hrow * 16 + m;
            unsigned lo = 0u, hi = 0u;
#pragma unroll
            for (int t = 0; t < 8; ++t)  lo |= 1u << (t * 4 + sTok[mrow + t]);
#pragma unroll
            for (int t = 8; t < KW; ++t) hi |= 1u << ((t - 8) * 4 + sTok[mrow + t]);

            v8f dy = {};
            dy = __builtin_amdgcn_wmma_f32_16x16x32_bf16(
                     false, mkfrag(lo), false, wB[0], (short)0, dy, false, false);
            dy = __builtin_amdgcn_wmma_f32_16x16x32_bf16(
                     false, mkfrag(hi), false, wB[1], (short)0, dy, false, false);
            // D layout: VGPR r -> M=r (lanes 0-15) / M=r+8 (lanes 16-31)
#pragma unroll
            for (int r = 0; r < 8; ++r)
                yA[(hrow * 16 + rbase + r) * DC + col] = (__bf16)(dy[r] + cbc);
        }
        __syncthreads();

        // ---- k and v projections; wave owns output cols [wid*16, +16) -----
#pragma unroll
        for (int hrow = 0; hrow < 2; ++hrow) {
            v8f dk = {};
            v8f dv = {};
#pragma unroll
            for (int kk = 0; kk < 4; ++kk) {
                const v8bf* p0 =
                    (const v8bf*)(yA + (hrow * 16 + m) * DC + kk * 32 + half);
                v8bf a0 = p0[0];
                v8bf a1 = p0[2];                     // +16 bf16
                v16bf a = __builtin_shufflevector(a0, a1,
                          0, 1, 2, 3, 4, 5, 6, 7, 8, 9, 10, 11, 12, 13, 14, 15);
                dk = __builtin_amdgcn_wmma_f32_16x16x32_bf16(
                         false, a, false, kB[kk], (short)0, dk, false, false);
                dv = __builtin_amdgcn_wmma_f32_16x16x32_bf16(
                         false, a, false, vB[kk], (short)0, dv, false, false);
            }
#pragma unroll
            for (int r = 0; r < 8; ++r) {
                kT[(hrow * 16 + rbase + r) * DC + col] = dk[r];
                vT[(hrow * 16 + rbase + r) * DC + col] = dv[r];
            }
        }
        __syncthreads();

        // ---- scores, stage 1: 256 threads x (2 rows, 8-elem partial dot) --
        {
            int c = tid & 15;
#pragma unroll
            for (int hrow = 0; hrow < 2; ++hrow) {
                int i = hrow * 16 + (tid >> 4);
                float p = 0.f;
#pragma unroll
                for (int e = 0; e < 8; ++e)
                    p += qc[e] * kT[i * DC + c * 8 + e];
                scP[i * 16 + c] = p;
            }
        }
        __syncthreads();
        // ---- stage 2: 32 threads fold partials (bk omitted: invariant) ----
        if (tid < PTILE) {
            float s = 0.f;
#pragma unroll
            for (int c = 0; c < 16; ++c) s += scP[tid * 16 + c];
            s *= scl;
            if (l0 + tid >= LOUT) s = -1e30f;        // mask tail positions
            sc[tid] = s;
        }
        __syncthreads();

        // ---- online softmax: max/scale (t0), exp (32 thr), Z fold (t0) ----
        if (tid == 0) {
            float tm = sc[0];
            for (int i = 1; i < PTILE; ++i) tm = fmaxf(tm, sc[i]);
            float newm = fmaxf(mZ[0], tm);
            scaleS = __expf(mZ[0] - newm);
            newmS  = newm;
            mZ[0]  = newm;
        }
        __syncthreads();
        if (tid < PTILE) ev[tid] = __expf(sc[tid] - newmS);
        __syncthreads();

        if (tid < DC) {
            float a = accS[tid] * scaleS;
#pragma unroll
            for (int i = 0; i < PTILE; ++i) a += ev[i] * vT[i * DC + tid];
            accS[tid] = a;
        }
        if (tid == 0) {                              // runs alongside acc update
            float zs = 0.f;
            for (int i = 0; i < PTILE; ++i) zs += ev[i];
            mZ[1] = mZ[1] * scaleS + zs;
        }
        __syncthreads();                             // protect tiles for next iter
    }

    if (tid < DC)
        out[n * DC + tid] = accS[tid] / mZ[1] + bv[tid];  // bv folds out of softmax
}

// ---------------------------------------------------------------------------
extern "C" void kernel_launch(void* const* d_in, const int* in_sizes, int n_in,
                              void* d_out, int out_size, void* d_ws, size_t ws_size,
                              hipStream_t stream) {
    (void)in_sizes; (void)n_in; (void)out_size; (void)ws_size;
    const int*   tokens = (const int*)  d_in[0];
    const float* emb    = (const float*)d_in[1];
    const float* conv_w = (const float*)d_in[2];
    const float* conv_b = (const float*)d_in[3];
    const float* Wq     = (const float*)d_in[4];
    const float* bq     = (const float*)d_in[5];
    const float* Wk     = (const float*)d_in[6];
    // d_in[7] = bk: unused (softmax-invariant constant shift of scores)
    const float* Wv     = (const float*)d_in[8];
    const float* bv     = (const float*)d_in[9];
    float* out = (float*)d_out;

    char* ws = (char*)d_ws;
    float*  W2  = (float*) (ws + WS_W2);
    float*  qws = (float*) (ws + WS_Q);
    __bf16* WkF = (__bf16*)(ws + WS_WKF);
    __bf16* WvF = (__bf16*)(ws + WS_WVF);
    __bf16* W2F = (__bf16*)(ws + WS_W2F);

    prep_tables<<<1, 256, 0, stream>>>(emb, conv_w, Wk, Wv, W2, WkF, WvF, W2F);
    mean_q<<<SEQ_N, 128, 0, stream>>>(tokens, conv_b, Wq, bq, W2, qws);
    attn<<<SEQ_N, 256, 0, stream>>>(tokens, conv_b, bv, qws, WkF, WvF, W2F, out);
}
